// MemoryEfficientAttention_1855425872543
// MI455X (gfx1250) — matrix operand order repaired and checked by reading, not compiled, not run on hardware
//
#include <hip/hip_runtime.h>
#include <hip/hip_bf16.h>
#include <math.h>

// Problem constants (B=1, S=4096, H=2048, 16 heads x 128)
#define SLEN   4096
#define HDIM   2048
#define NHEADS 16
#define HEADD  128
#define ATT_SCALE 0.08838834764831845f   // 1/sqrt(128)

typedef __attribute__((ext_vector_type(16))) __bf16 bf16x16;
typedef __attribute__((ext_vector_type(8)))  __bf16 bf16x8;
typedef __attribute__((ext_vector_type(8)))  float  f32x8;
typedef __attribute__((ext_vector_type(4)))  float  f32x4;

static __device__ __forceinline__ f32x8 wmma_bf16(bf16x16 a, bf16x16 b, f32x8 c) {
  // D = A(16x32 bf16) * B(32x16 bf16) + C(16x16 f32)
  return __builtin_amdgcn_wmma_f32_16x16x32_bf16(false, a, false, b, (short)0, c,
                                                 false, false);
}

// ---- Fragment loaders (CDNA5 ISA 7.12.2 layouts) -------------------------
// A-matrix 16x32 (MxK), 16-bit: lane L<16 -> row M=L, K={0..7} in v0-3,
// K={16..23} in v4-7; lane L+16 -> same row, K={8..15},{24..31}.
static __device__ __forceinline__ bf16x16 load_a_bf16(const __bf16* base, int ld) {
  const int lane = threadIdx.x & 31;
  const __bf16* p = base + (size_t)(lane & 15) * ld + ((lane >> 4) << 3);
  bf16x16 f;
  ((uint4*)&f)[0] = *(const uint4*)(p);        // K ko .. ko+7
  ((uint4*)&f)[1] = *(const uint4*)(p + 16);   // K ko+16 .. ko+23
  return f;
}

// B-matrix 32x16 (KxN), 16-bit: lane L<16 -> col N=L, K={0..7},{16..23};
// lane L+16 -> col N=L, K={8..15},{24..31}.  Source is [N,K] row-major.
static __device__ __forceinline__ bf16x16 load_b_bf16(const __bf16* base, int ld) {
  const int lane = threadIdx.x & 31;
  const __bf16* p = base + (size_t)(lane & 15) * ld + ((lane >> 4) << 3);
  bf16x16 f;
  ((uint4*)&f)[0] = *(const uint4*)(p);        // K ko .. ko+7
  ((uint4*)&f)[1] = *(const uint4*)(p + 16);   // K ko+16 .. ko+23
  return f;
}

// ---- fp32 -> bf16 bulk convert (one-time; removes cvt from GEMM loops) ---
__global__ __launch_bounds__(256)
void mea_cvt_bf16(const float* __restrict__ in, __bf16* __restrict__ out) {
  const size_t i = ((size_t)blockIdx.x * blockDim.x + threadIdx.x) * 8;
  f32x4 a0 = *(const f32x4*)(in + i);
  f32x4 a1 = *(const f32x4*)(in + i + 4);
  bf16x8 v;
#pragma unroll
  for (int j = 0; j < 4; ++j) { v[j] = (__bf16)a0[j]; v[4 + j] = (__bf16)a1[j]; }
  *(bf16x8*)(out + i) = v;
}

// ---- Projection GEMM: out = (A * B^T + bias) * oscale, bf16 output ------
// Wave tile 64x64 (4x4 WMMA tiles); block = 8 waves (2M x 4N) -> 128x256.
// TRANS: store out[n][s] (for V -> Vt, packed b128 stores) else out[s][n].
template <bool TRANS>
__global__ __launch_bounds__(256)
void mea_proj_gemm(const __bf16* __restrict__ A, const __bf16* __restrict__ B,
                   const float* __restrict__ bias, __bf16* __restrict__ out,
                   int M, int N, int K, float oscale) {
  const int wid   = threadIdx.x >> 5;
  const int mbase = blockIdx.y * 128 + (wid & 1) * 64;
  const int nbase = blockIdx.x * 256 + (wid >> 1) * 64;

  f32x8 acc[4][4];
#pragma unroll
  for (int mi = 0; mi < 4; ++mi)
#pragma unroll
    for (int ni = 0; ni < 4; ++ni) acc[mi][ni] = (f32x8){};

  for (int k = 0; k < K; k += 32) {
    bf16x16 af[4], bfr[4];
#pragma unroll
    for (int mi = 0; mi < 4; ++mi) {
      const __bf16* pa = A + (size_t)(mbase + 16 * mi) * K + k;
      af[mi] = load_a_bf16(pa, K);
      __builtin_prefetch(pa + 256, 0, 1);      // global_prefetch_b8, 512B ahead
    }
#pragma unroll
    for (int ni = 0; ni < 4; ++ni)
      bfr[ni] = load_b_bf16(B + (size_t)(nbase + 16 * ni) * K + k, K);
#pragma unroll
    for (int mi = 0; mi < 4; ++mi)
#pragma unroll
      for (int ni = 0; ni < 4; ++ni)
        acc[mi][ni] = wmma_bf16(af[mi], bfr[ni], acc[mi][ni]);
  }

  // C/D layout: VGPR i, lanes0-15 -> (M=i, N=lane); lanes16-31 -> (M=i+8).
  const int lane = threadIdx.x & 31;
  const int hi = lane >> 4, ln = lane & 15;
#pragma unroll
  for (int ni = 0; ni < 4; ++ni) {
    const int col = nbase + 16 * ni + ln;
    const float bv = bias[col];
#pragma unroll
    for (int mi = 0; mi < 4; ++mi) {
      if (TRANS) {
        // 8 consecutive rows of one column -> one packed 16B store to Vt[n][s]
        bf16x8 v;
#pragma unroll
        for (int i = 0; i < 8; ++i) v[i] = (__bf16)((acc[mi][ni][i] + bv) * oscale);
        *(bf16x8*)(out + (size_t)col * M + mbase + 16 * mi + 8 * hi) = v;
      } else {
#pragma unroll
        for (int i = 0; i < 8; ++i) {
          const int row = mbase + 16 * mi + i + 8 * hi;
          out[(size_t)row * N + col] = (__bf16)((acc[mi][ni][i] + bv) * oscale);
        }
      }
    }
  }
}

// ---- Flash attention: one wave = 16 query rows x full head-dim -----------
__global__ __launch_bounds__(128)
void mea_flash_attn(const __bf16* __restrict__ Q, const __bf16* __restrict__ Km,
                    const __bf16* __restrict__ Vt, __bf16* __restrict__ O) {
  const int head  = blockIdx.y;
  const int wid   = threadIdx.x >> 5;
  const int qbase = blockIdx.x * 64 + wid * 16;
  const int lane  = threadIdx.x & 31;
  const int hi = lane >> 4, ln = lane & 15;

  __shared__ __align__(16) __bf16 pshare[4][16 * 48];   // 48-elem row stride
  __bf16* plds = pshare[wid];

  // Q fragments (already scaled by 1/sqrt(d)) stay in registers throughout.
  const __bf16* qptr = Q + (size_t)qbase * HDIM + head * HEADD;
  bf16x16 qf[4];
#pragma unroll
  for (int c = 0; c < 4; ++c) qf[c] = load_a_bf16(qptr + c * 32, HDIM);

  f32x8 acc[8];
#pragma unroll
  for (int t = 0; t < 8; ++t) acc[t] = (f32x8){};
  float m[8], l[8];
#pragma unroll
  for (int i = 0; i < 8; ++i) { m[i] = -INFINITY; l[i] = 0.f; }

  for (int kb = 0; kb < SLEN; kb += 32) {
    // scores: two 16x16 tiles over key columns kb..kb+31
    const __bf16* kp = Km + (size_t)kb * HDIM + head * HEADD;
    f32x8 s0 = (f32x8){}, s1 = (f32x8){};
#pragma unroll
    for (int c = 0; c < 4; ++c) {
      bf16x16 b0 = load_b_bf16(kp + c * 32, HDIM);
      s0 = wmma_bf16(qf[c], b0, s0);
    }
#pragma unroll
    for (int c = 0; c < 4; ++c) {
      bf16x16 b1 = load_b_bf16(kp + (size_t)16 * HDIM + c * 32, HDIM);
      s1 = wmma_bf16(qf[c], b1, s1);
    }

    // Online softmax per row (row stats reduced across the 16-lane half).
#pragma unroll
    for (int i = 0; i < 8; ++i) {
      const float a = s0[i];
      const float b = s1[i];
      float r = fmaxf(a, b);
      r = fmaxf(r, __shfl_xor(r, 1, 32));
      r = fmaxf(r, __shfl_xor(r, 2, 32));
      r = fmaxf(r, __shfl_xor(r, 4, 32));
      r = fmaxf(r, __shfl_xor(r, 8, 32));
      const float mn    = fmaxf(m[i], r);
      const float alpha = __expf(m[i] - mn);
      m[i] = mn;
      const float p0 = __expf(a - mn);
      const float p1 = __expf(b - mn);
      float rs = p0 + p1;
      rs += __shfl_xor(rs, 1, 32);
      rs += __shfl_xor(rs, 2, 32);
      rs += __shfl_xor(rs, 4, 32);
      rs += __shfl_xor(rs, 8, 32);
      l[i] = l[i] * alpha + rs;
#pragma unroll
      for (int t = 0; t < 8; ++t) acc[t][i] *= alpha;
      // stage P (C layout -> row-major bf16) in per-wave LDS
      const int row = i + 8 * hi;
      plds[row * 48 + ln]      = (__bf16)p0;
      plds[row * 48 + 16 + ln] = (__bf16)p1;
    }

    // Re-load P as an A-fragment (LDS ops in-order within a wave).
    bf16x16 pf;
    {
      const __bf16* pp = plds + ln * 48 + hi * 8;
      ((uint4*)&pf)[0] = *(const uint4*)(pp);
      ((uint4*)&pf)[1] = *(const uint4*)(pp + 16);
    }

    // out += P (16x32) * V (32k x 16d) per 16-wide d tile; Vt[n][s] layout
    // makes the B fragment a contiguous load along s.
    const __bf16* vp = Vt + (size_t)head * HEADD * SLEN + kb;
#pragma unroll
    for (int t = 0; t < 8; ++t) {
      bf16x16 bv = load_b_bf16(vp + (size_t)(t * 16) * SLEN, SLEN);
      acc[t] = wmma_bf16(pf, bv, acc[t]);
    }
  }

  // epilogue: normalize and store bf16 attention output [s][head*128+d]
  __bf16* optr = O + (size_t)qbase * HDIM + head * HEADD;
#pragma unroll
  for (int i = 0; i < 8; ++i) {
    const float inv = 1.0f / l[i];
    const int row = i + 8 * hi;
#pragma unroll
    for (int t = 0; t < 8; ++t)
      optr[(size_t)row * HDIM + t * 16 + ln] = (__bf16)(acc[t][i] * inv);
  }
}

// ---- Output projection: Y = O * Wo^T + bo, fp32 output -------------------
__global__ __launch_bounds__(256)
void mea_out_proj(const __bf16* __restrict__ A, const __bf16* __restrict__ B,
                  const float* __restrict__ bias, float* __restrict__ out,
                  int M, int N, int K) {
  const int wid   = threadIdx.x >> 5;
  const int mbase = blockIdx.y * 128 + (wid & 1) * 64;
  const int nbase = blockIdx.x * 256 + (wid >> 1) * 64;

  f32x8 acc[4][4];
#pragma unroll
  for (int mi = 0; mi < 4; ++mi)
#pragma unroll
    for (int ni = 0; ni < 4; ++ni) acc[mi][ni] = (f32x8){};

  for (int k = 0; k < K; k += 32) {
    bf16x16 af[4], bfr[4];
#pragma unroll
    for (int mi = 0; mi < 4; ++mi) {
      const __bf16* pa = A + (size_t)(mbase + 16 * mi) * K + k;
      af[mi] = load_a_bf16(pa, K);
      __builtin_prefetch(pa + 256, 0, 1);
    }
#pragma unroll
    for (int ni = 0; ni < 4; ++ni)
      bfr[ni] = load_b_bf16(B + (size_t)(nbase + 16 * ni) * K + k, K);
#pragma unroll
    for (int mi = 0; mi < 4; ++mi)
#pragma unroll
      for (int ni = 0; ni < 4; ++ni)
        acc[mi][ni] = wmma_bf16(af[mi], bfr[ni], acc[mi][ni]);
  }

  const int lane = threadIdx.x & 31;
  const int hi = lane >> 4, ln = lane & 15;
#pragma unroll
  for (int ni = 0; ni < 4; ++ni) {
    const int col = nbase + 16 * ni + ln;
    const float bv = bias[col];
#pragma unroll
    for (int mi = 0; mi < 4; ++mi)
#pragma unroll
      for (int i = 0; i < 8; ++i) {
        const int row = mbase + 16 * mi + i + 8 * hi;
        out[(size_t)row * N + col] = acc[mi][ni][i] + bv;
      }
  }
}

extern "C" void kernel_launch(void* const* d_in, const int* in_sizes, int n_in,
                              void* d_out, int out_size, void* d_ws, size_t ws_size,
                              hipStream_t stream) {
  (void)in_sizes; (void)n_in; (void)out_size; (void)ws_size;
  const float* X  = (const float*)d_in[0];
  const float* Wq = (const float*)d_in[1];
  const float* bq = (const float*)d_in[2];
  const float* Wk = (const float*)d_in[3];
  const float* bk = (const float*)d_in[4];
  const float* Wv = (const float*)d_in[5];
  const float* bv = (const float*)d_in[6];
  const float* Wo = (const float*)d_in[7];
  const float* bo = (const float*)d_in[8];
  float* Y = (float*)d_out;

  // Workspace layout (bf16, all L2-resident: 112 MB total)
  const size_t xe = (size_t)SLEN * HDIM;               // 8M elements
  const size_t we = (size_t)HDIM * HDIM;               // 4M elements
  __bf16* Xb  = (__bf16*)d_ws;
  __bf16* Wqb = Xb  + xe;
  __bf16* Wkb = Wqb + we;
  __bf16* Wvb = Wkb + we;
  __bf16* Wob = Wvb + we;
  __bf16* Qb  = Wob + we;
  __bf16* Kb  = Qb  + xe;
  __bf16* Vtb = Kb  + xe;                              // [NH*HD][S] transposed
  __bf16* Ob  = Vtb + xe;

  // one-time fp32 -> bf16 conversions (8 elems/thread)
  mea_cvt_bf16<<<dim3(xe / (8 * 256)), dim3(256), 0, stream>>>(X,  Xb);
  mea_cvt_bf16<<<dim3(we / (8 * 256)), dim3(256), 0, stream>>>(Wq, Wqb);
  mea_cvt_bf16<<<dim3(we / (8 * 256)), dim3(256), 0, stream>>>(Wk, Wkb);
  mea_cvt_bf16<<<dim3(we / (8 * 256)), dim3(256), 0, stream>>>(Wv, Wvb);
  mea_cvt_bf16<<<dim3(we / (8 * 256)), dim3(256), 0, stream>>>(Wo, Wob);

  dim3 gblk(256);
  dim3 ggrid(HDIM / 256, SLEN / 128);                  // (8, 32)
  // Q is pre-scaled by 1/sqrt(head_dim) so attention skips the logit scale.
  mea_proj_gemm<false><<<ggrid, gblk, 0, stream>>>(Xb, Wqb, bq, Qb,  SLEN, HDIM, HDIM, ATT_SCALE);
  mea_proj_gemm<false><<<ggrid, gblk, 0, stream>>>(Xb, Wkb, bk, Kb,  SLEN, HDIM, HDIM, 1.0f);
  mea_proj_gemm<true ><<<ggrid, gblk, 0, stream>>>(Xb, Wvb, bv, Vtb, SLEN, HDIM, HDIM, 1.0f);

  dim3 ablk(128);
  dim3 agrid(SLEN / 64, NHEADS);                       // (64, 16)
  mea_flash_attn<<<agrid, ablk, 0, stream>>>(Qb, Kb, Vtb, Ob);

  mea_out_proj<<<ggrid, gblk, 0, stream>>>(Ob, Wob, bo, Y, SLEN, HDIM, HDIM);
}
